// Transolver_300647710952
// MI455X (gfx1250) — matrix operand order repaired and checked by reading, not compiled, not run on hardware
//
#include <hip/hip_runtime.h>

typedef __attribute__((ext_vector_type(16))) _Float16 v16h;
typedef __attribute__((ext_vector_type(8)))  float    v8f;

#define BATCH  2
#define HDIM   128
#define WDIM   128
#define TWD    5
#define NTOT   (TWD*HDIM*WDIM)   /* 81920 */
#define NHEADS 8
#define DHD    64
#define GSL    32
#define KRAW   189
#define KPAD   192               /* 189 conv taps + 1 bias row + 2 zero */
#define TILE_N 64

// ---- fragment index helpers (16-bit operands, 16x16x32; contiguous k = 16*half + elem) ----
// A (MxK): forward map lane/elem -> K
__device__ __forceinline__ int aKmap(int lane, int e) {
  int r = e >> 1, h = e & 1;
  int base = (lane < 16) ? 0 : 16;
  return base + 2 * r + h;
}
// A inverse: K-within-tile (kk in [0,32)) -> lane-half offset (0|16) and elem
__device__ __forceinline__ int aInvLane(int kk) { return (kk & 16) ? 16 : 0; }
__device__ __forceinline__ int aInvElem(int kk) { return kk & 15; }
// B (KxN): lanes 0-15 hold K=0..15, lanes 16-31 hold K=16..31 (elem = kk&15)

__global__ void k_zero(float* __restrict__ p, int n) {
  int i = blockIdx.x * 256 + threadIdx.x;
  if (i < n) p[i] = 0.0f;
}

// ---------------------------------------------------------------------------
// K1: fused conv3d(x2) -> slice logits -> softmax -> slice_norm/slice_token
// grid.x = BATCH * NHEADS * (NTOT/TILE_N), block = 256 (8 waves)
// Conv A/B and x_mid/fx_mid B operands staged in fragment-major LDS layout: [tile][kt][lane]v16h
// ---------------------------------------------------------------------------
__global__ void __launch_bounds__(256)
k_conv_slice(const float* __restrict__ x,
             const float* __restrict__ cxw, const float* __restrict__ cxb,
             const float* __restrict__ cfw, const float* __restrict__ cfb,
             const float* __restrict__ slw, const float* __restrict__ slb,
             const float* __restrict__ temp,
             _Float16* __restrict__ swg,    // [B][N][256] slice weights (f16)
             float* __restrict__ tok,       // [B*H][32][64] accum (f32)
             float* __restrict__ nrm)       // [B*H][32] accum (f32)
{
  // 48KB overlaid pool + 2x8KB persistent fragment arrays = 64KB
  __shared__ __align__(32) unsigned char pool[49152];
  _Float16* aFh = (_Float16*)pool;                  // conv A frags: [4 mt][6 kt][32][16]
  _Float16* bFh = (_Float16*)(pool + 24576);        // conv B frags: [4 nt][6 kt][32][16]
  const v16h* aF = (const v16h*)pool;
  const v16h* bF = (const v16h*)(pool + 24576);
  float    (*lgf)[TILE_N] = (float(*)[TILE_N])pool;           // [32][64] (aliases aF)
  _Float16 (*sw)[TILE_N]  = (_Float16(*)[TILE_N])(pool + 8192);  // [32][64]
  _Float16 (*swl)[DHD]    = (_Float16(*)[DHD])(pool + 12288);    // [32][64]
  float*    sbias         = (float*)(pool + 16384);              // [32]
  __shared__ __align__(32) _Float16 xmFh[4 * 2 * 32 * 16];   // x_mid  B-frags [nt][kt][lane][16]
  __shared__ __align__(32) _Float16 fxmFh[4 * 2 * 32 * 16];  // fx_mid B-frags
  const v16h* xmF  = (const v16h*)xmFh;
  const v16h* fxmF = (const v16h*)fxmFh;

  const int tid  = threadIdx.x;
  const int wid  = tid >> 5;
  const int lane = tid & 31;
  const int lm   = lane & 15;
  const int hi8  = (lane >> 4) << 3;   // C/D row offset for upper half-wave

  const int ntile = NTOT / TILE_N;
  int bid  = blockIdx.x;
  int tile = bid % ntile;
  int bh   = bid / ntile;
  int hd   = bh & 7, bb = bh >> 3;
  int n0   = tile * TILE_N;
  int z    = n0 / (HDIM * WDIM);
  int rem  = n0 % (HDIM * WDIM);
  int yy0  = rem / WDIM;
  int x0   = rem % WDIM;

  // ---- stage input patch straight into B-fragment layout (shared by both convs)
  for (int e = tid; e < KPAD * TILE_N; e += 256) {
    int k = e >> 6, p = e & 63;            // p fastest -> coalesced global reads
    _Float16 val = (_Float16)0.0f;
    if (k < KRAW) {
      int ic = k / 27, r = k % 27;
      int dz = r / 9; r %= 9;
      int dy = r / 3, dx = r % 3;
      int zz = z + dz - 1, yv = yy0 + dy - 1, xv = x0 + p + dx - 1;
      if (zz >= 0 && zz < TWD && yv >= 0 && yv < HDIM && xv >= 0 && xv < WDIM) {
        int ch;
        switch (ic) {                 // x5 channel -> source channel in x[:,27]
          case 0:  ch = zz;      break;   // temp
          case 1:  ch = 5 + zz;  break;   // u
          case 2:  ch = 15 + zz; break;   // v
          case 3:  ch = 25;      break;   // hc (broadcast over z)
          case 4:  ch = 26;      break;   // wc (broadcast over z)
          case 5:  ch = 10 + zz; break;   // u_f
          default: ch = 20 + zz; break;   // v_f
        }
        val = (_Float16)x[(((size_t)bb * 27 + ch) * HDIM + yv) * WDIM + xv];
      }
    } else if (k == KRAW) {
      val = (_Float16)1.0f;              // bias row: patch==1, weight==bias
    }
    int nt = p >> 4, kt = k >> 5, kk = k & 31;
    int lane2 = (p & 15) + ((kk >= 16) ? 16 : 0);
    bFh[(((nt * 6 + kt) * 32 + lane2) << 4) + (kk & 15)] = val;
  }

  // ---- two convs as implicit GEMM: M=64 out-ch, K=192 (taps+bias), N=64 pos
#pragma unroll
  for (int cc = 0; cc < 2; ++cc) {
    const float* wsrc = cc ? cfw : cxw;
    const float* bsrc = cc ? cfb : cxb;
    __syncthreads();
    // stage this head's weights (+bias in row 189) into A-fragment layout
    for (int e = tid; e < DHD * KPAD; e += 256) {
      int oc = e / KPAD, k = e % KPAD;   // k fastest -> coalesced global reads
      float v;
      if (k < KRAW)       v = wsrc[(size_t)(hd * DHD + oc) * KRAW + k];
      else if (k == KRAW) v = bsrc[hd * DHD + oc];
      else                v = 0.0f;
      int mt = oc >> 4, kt = k >> 5, kk = k & 31;
      int lane2 = (oc & 15) + aInvLane(kk);
      aFh[(((mt * 6 + kt) * 32 + lane2) << 4) + aInvElem(kk)] = (_Float16)v;
    }
    __syncthreads();
    // 16 tile jobs (4 m-tiles x 4 n-tiles): exactly 2 per wave, fully unrolled
    // so no divergent-loop EXEC machinery around the WMMAs.
#pragma unroll
    for (int jj = 0; jj < 2; ++jj) {
      int job = wid + jj * 8;
      int mt = job >> 2, nt = job & 3;
      v8f acc = {};
#pragma unroll
      for (int kt = 0; kt < 6; ++kt) {
        v16h a = aF[(mt * 6 + kt) * 32 + lane];     // one 32B read per operand
        v16h b = bF[(nt * 6 + kt) * 32 + lane];
        acc = __builtin_amdgcn_wmma_f32_16x16x32_f16(false, a, false, b, (short)0, acc, false, false);
      }
#pragma unroll
      for (int r2 = 0; r2 < 8; ++r2) {
        int oc = mt * 16 + r2 + hi8;
        int p  = nt * 16 + lm;
        _Float16 hv = (_Float16)acc[r2];
        if (cc == 0) {   // x_mid as B-frags for logits GEMM: K=oc(c), N=p(pos)
          int kt2 = oc >> 5, kk = oc & 31;
          int lane2 = (p & 15) + ((kk >= 16) ? 16 : 0);
          xmFh[((((p >> 4) * 2 + kt2) * 32 + lane2) << 4) + (kk & 15)] = hv;
        } else {         // fx_mid as B-frags for token GEMM: K=p(pos), N=oc(c)
          int kt2 = p >> 5, kk = p & 31;
          int lane2 = (oc & 15) + ((kk >= 16) ? 16 : 0);
          fxmFh[((((oc >> 4) * 2 + kt2) * 32 + lane2) << 4) + (kk & 15)] = hv;
        }
      }
    }
  }
  __syncthreads();

  // ---- stage slice_w / slice_b over dead conv-fragment region ----
  for (int e = tid; e < GSL * DHD; e += 256) swl[e / DHD][e % DHD] = (_Float16)slw[e];
  if (tid < GSL) sbias[tid] = slb[tid];
  __syncthreads();

  // ---- logits(G x pos) = slice_w(GxDH) @ x_mid(DHxpos); one tile per wave ----
  {
    int mt = wid >> 2, nt = wid & 3;   // 2 g-tiles x 4 pos-tiles
    v8f acc = {};
#pragma unroll
    for (int kt = 0; kt < 2; ++kt) {
      v16h a, b;
#pragma unroll
      for (int e2 = 0; e2 < 16; ++e2)
        a[e2] = swl[mt * 16 + lm][kt * 32 + aKmap(lane, e2)];
      b = xmF[(nt * 2 + kt) * 32 + lane];
      acc = __builtin_amdgcn_wmma_f32_16x16x32_f16(false, a, false, b, (short)0, acc, false, false);
    }
#pragma unroll
    for (int r2 = 0; r2 < 8; ++r2) {
      int g = mt * 16 + r2 + hi8;
      lgf[g][nt * 16 + lm] = acc[r2] + sbias[g];
    }
  }
  __syncthreads();

  // ---- softmax over G per position, / clipped temperature ----
  {
    float tc  = fminf(fmaxf(temp[hd], 0.1f), 5.0f);
    float itc = 1.0f / tc;
    if (tid < TILE_N) {
      int p = tid;
      float mx = -3.0e38f;
      for (int g = 0; g < GSL; ++g) mx = fmaxf(mx, lgf[g][p] * itc);
      float s = 0.0f;
      for (int g = 0; g < GSL; ++g) { float e = __expf(lgf[g][p] * itc - mx); lgf[g][p] = e; s += e; }
      float inv = 1.0f / s;
      for (int g = 0; g < GSL; ++g) sw[g][p] = (_Float16)(lgf[g][p] * inv);
    }
  }
  __syncthreads();

  // ---- slice_norm partial sums + persist slice weights ([b][n][h*32+g]) ----
  if (tid < GSL) {
    float s = 0.0f;
    for (int p = 0; p < TILE_N; ++p) s += (float)sw[tid][p];
    atomicAdd(&nrm[(bb * NHEADS + hd) * GSL + tid], s);
  }
  for (int e = tid; e < TILE_N * GSL; e += 256) {
    int p = e >> 5, g = e & 31;
    swg[((size_t)bb * NTOT + n0 + p) * 256 + hd * GSL + g] = sw[g][p];
  }

  // ---- slice_token partial: (G x pos) @ (pos x C) -> global f32 atomics ----
  {
    int mt = wid >> 2, nt = wid & 3;   // 2 g-tiles x 4 c-tiles
    v8f acc = {};
#pragma unroll
    for (int kt = 0; kt < 2; ++kt) {
      v16h a, b;
#pragma unroll
      for (int e2 = 0; e2 < 16; ++e2)
        a[e2] = sw[mt * 16 + lm][kt * 32 + aKmap(lane, e2)];
      b = fxmF[(nt * 2 + kt) * 32 + lane];
      acc = __builtin_amdgcn_wmma_f32_16x16x32_f16(false, a, false, b, (short)0, acc, false, false);
    }
#pragma unroll
    for (int r2 = 0; r2 < 8; ++r2) {
      int g = mt * 16 + r2 + hi8;
      int c = nt * 16 + lm;
      atomicAdd(&tok[((size_t)(bb * NHEADS + hd) * GSL + g) * DHD + c], acc[r2]);
    }
  }
}

// ---------------------------------------------------------------------------
// K2: per (b,h): normalize token, q/k/v, 32x32 attention, collapse with out_w
// ---------------------------------------------------------------------------
__global__ void __launch_bounds__(256)
k_attn(const float* __restrict__ wq, const float* __restrict__ wk,
       const float* __restrict__ wvv, const float* __restrict__ outw,
       const float* __restrict__ tok, const float* __restrict__ nrm,
       float* __restrict__ sarr)
{
  __shared__ float tk[GSL][DHD];
  __shared__ float qm[GSL][DHD];
  __shared__ float km[GSL][DHD];
  __shared__ float vm[GSL][DHD];
  __shared__ float at[GSL][GSL];
  int tid  = threadIdx.x;
  int bhid = blockIdx.x;         // bb*8 + hd
  int hd   = bhid & 7;
  const float* tsrc = tok + (size_t)bhid * GSL * DHD;
  const float* nsrc = nrm + (size_t)bhid * GSL;

  for (int e = tid; e < GSL * DHD; e += 256) {
    int g = e >> 6, c = e & 63;
    tk[g][c] = tsrc[e] / (nsrc[g] + 1e-5f);
  }
  __syncthreads();
  for (int e = tid; e < GSL * DHD; e += 256) {
    int g = e >> 6, c = e & 63;
    float aq = 0.f, ak = 0.f, av = 0.f;
    for (int j = 0; j < DHD; ++j) {
      float t = tk[g][j];
      aq += t * wq[c * DHD + j];
      ak += t * wk[c * DHD + j];
      av += t * wvv[c * DHD + j];
    }
    qm[g][c] = aq; km[g][c] = ak; vm[g][c] = av;
  }
  __syncthreads();
  for (int e = tid; e < GSL * GSL; e += 256) {
    int g = e >> 5, g2 = e & 31;
    float s = 0.f;
    for (int c = 0; c < DHD; ++c) s += qm[g][c] * km[g2][c];
    at[g][g2] = s * 0.125f;      // DH^-0.5
  }
  __syncthreads();
  if (tid < GSL) {
    float mx = -3.0e38f;
    for (int g2 = 0; g2 < GSL; ++g2) mx = fmaxf(mx, at[tid][g2]);
    float s = 0.f;
    for (int g2 = 0; g2 < GSL; ++g2) { float e = __expf(at[tid][g2] - mx); at[tid][g2] = e; s += e; }
    float inv = 1.0f / s;
    for (int g2 = 0; g2 < GSL; ++g2) at[tid][g2] *= inv;
  }
  __syncthreads();
  for (int e = tid; e < GSL * DHD; e += 256) {
    int g = e >> 6, c = e & 63;
    float s = 0.f;
    for (int g2 = 0; g2 < GSL; ++g2) s += at[g][g2] * vm[g2][c];
    qm[g][c] = s;                // reuse qm as out_tok
  }
  __syncthreads();
  if (tid < GSL) {
    float s = 0.f;
    for (int c = 0; c < DHD; ++c) s += qm[tid][c] * outw[hd * DHD + c];
    sarr[bhid * GSL + tid] = s;  // s(b,h,g) = out_tok . out_w segment
  }
}

// ---------------------------------------------------------------------------
// K3: out(b,n) = out_b + sum_{h,g} w(b,n,h,g) * s(b,h,g)   (streams 84MB f16)
// ---------------------------------------------------------------------------
__global__ void __launch_bounds__(256)
k_out(const _Float16* __restrict__ swg, const float* __restrict__ sarr,
      const float* __restrict__ outb, float* __restrict__ out)
{
  __shared__ float ss[256];
  int tid = threadIdx.x;
  size_t idx = (size_t)blockIdx.x * 256 + tid;
  int bb = (int)(idx / NTOT);          // 256 | NTOT -> uniform per block
  ss[tid] = sarr[bb * 256 + tid];
  __syncthreads();
  const v16h* row = (const v16h*)(swg + idx * 256);
  float acc = outb[0];
  for (int i = 0; i < 16; ++i) {       // 16 x 32B loads per position
    v16h h = row[i];
#pragma unroll
    for (int j = 0; j < 16; ++j) acc += (float)h[j] * ss[i * 16 + j];
  }
  out[idx] = acc;
}

extern "C" void kernel_launch(void* const* d_in, const int* in_sizes, int n_in,
                              void* d_out, int out_size, void* d_ws, size_t ws_size,
                              hipStream_t stream)
{
  (void)in_sizes; (void)n_in; (void)out_size; (void)ws_size;
  const float* x    = (const float*)d_in[0];
  const float* cxw  = (const float*)d_in[1];
  const float* cxb  = (const float*)d_in[2];
  const float* cfw  = (const float*)d_in[3];
  const float* cfb  = (const float*)d_in[4];
  const float* slw  = (const float*)d_in[5];
  const float* slb  = (const float*)d_in[6];
  const float* temp = (const float*)d_in[7];
  const float* wq   = (const float*)d_in[8];
  const float* wk   = (const float*)d_in[9];
  const float* wvv  = (const float*)d_in[10];
  const float* outw = (const float*)d_in[11];
  const float* outb = (const float*)d_in[12];

  const size_t swBytes  = (size_t)BATCH * NTOT * 256 * sizeof(_Float16); // 83,886,080
  const size_t tokBytes = (size_t)BATCH * NHEADS * GSL * DHD * sizeof(float);
  const size_t nrmBytes = (size_t)BATCH * NHEADS * GSL * sizeof(float);
  char* ws = (char*)d_ws;
  _Float16* swg = (_Float16*)ws;
  float* tok  = (float*)(ws + swBytes);
  float* nrm  = (float*)(ws + swBytes + tokBytes);
  float* sarr = (float*)(ws + swBytes + tokBytes + nrmBytes);

  int nz = (int)((tokBytes + nrmBytes) / sizeof(float));  // tok+nrm contiguous
  k_zero<<<(nz + 255) / 256, 256, 0, stream>>>(tok, nz);
  k_conv_slice<<<BATCH * NHEADS * (NTOT / TILE_N), 256, 0, stream>>>(
      x, cxw, cxb, cfw, cfb, slw, slb, temp, swg, tok, nrm);
  k_attn<<<BATCH * NHEADS, 256, 0, stream>>>(wq, wk, wvv, outw, tok, nrm, sarr);
  k_out<<<(BATCH * NTOT) / 256, 256, 0, stream>>>(swg, sarr, outb, (float*)d_out);
}
